// AttnDownBlock_20950850470529
// MI455X (gfx1250) — matrix operand order repaired and checked by reading, not compiled, not run on hardware
//
#include <hip/hip_runtime.h>
#include <hip/hip_bf16.h>

typedef __attribute__((ext_vector_type(16))) _Float16 v16h;
typedef __attribute__((ext_vector_type(8)))  float    v8f;
typedef _Float16 f16;

union AFrag { v16h v; uint4 q[2]; f16 e[16]; };
union CFrag { v8f  v; float e[8]; };

__device__ __forceinline__ v8f wmma_f16(v16h a, v16h b, v8f c) {
  return __builtin_amdgcn_wmma_f32_16x16x32_f16(false, a, false, b, (short)0, c, false, false);
}

__device__ __forceinline__ float silu_f(float v) { return v / (1.f + __expf(-v)); }

// ---------------- NCHW f32 -> NHWC f32 + f16 ----------------
__global__ void k_nchw_to_nhwc(const float* __restrict__ x, float* __restrict__ xn,
                               f16* __restrict__ x16, int B, int C, int HW) {
  size_t idx = (size_t)blockIdx.x * blockDim.x + threadIdx.x;
  size_t total = (size_t)B * C * HW;
  if (idx >= total) return;
  int hw = (int)(idx % HW); size_t t = idx / HW;
  int c = (int)(t % C); int b = (int)(t / C);
  float v = x[idx];
  size_t o = ((size_t)b * HW + hw) * C + c;
  xn[o] = v; x16[o] = (f16)v;
}

// ---------------- time embedding ----------------
__global__ void k_temb(const float* __restrict__ emb, const float* __restrict__ tw,
                       const float* __restrict__ tb, float* __restrict__ out, int T, int Cout) {
  int c = blockIdx.x * blockDim.x + threadIdx.x;
  int b = blockIdx.y;
  if (c >= Cout) return;
  float acc = tb[c];
  for (int k = 0; k < T; ++k) {
    float e = emb[(size_t)b * T + k];
    acc += silu_f(e) * tw[(size_t)c * T + k];
  }
  out[(size_t)b * Cout + c] = acc;
}

// ---------------- weight pack into WMMA B-fragment order ----------------
// wf[(((kblk*ntiles)+ntile)*32 + lane)*16 + h] = w[co][ci][tap]
__global__ void k_pack_wf(const float* __restrict__ w, f16* __restrict__ wf,
                          int Cout, int Cin, int taps) {
  size_t i = (size_t)blockIdx.x * blockDim.x + threadIdx.x;
  size_t total = (size_t)Cout * Cin * taps;
  if (i >= total) return;
  int h = (int)(i & 15);
  int lane = (int)((i >> 4) & 31);
  size_t r = i >> 9;
  int ntiles = Cout >> 4;
  int ntile = (int)(r % ntiles);
  int kblk = (int)(r / ntiles);
  int kk = kblk * 32 + ((h >> 3) * 16) + ((lane >> 4) * 8) + (h & 7);
  int tap = kk / Cin;
  int ci = kk - tap * Cin;
  int co = ntile * 16 + (lane & 15);
  wf[i] = (f16)w[((size_t)co * Cin + ci) * taps + tap];
}

// ---------------- GroupNorm (+opt SiLU) f32 NHWC -> f16 NHWC (opt zero-haloed pad) ----------------
__global__ void __launch_bounds__(256)
k_gn(const float* __restrict__ in, const float* __restrict__ sc, const float* __restrict__ bi,
     f16* __restrict__ out, int H, int W, int C, int Cg, int pad, int dosilu) {
  int b = blockIdx.x, g = blockIdx.y;
  int HWn = H * W;
  int n = HWn * Cg;
  const float* base = in + (size_t)b * HWn * C + (size_t)g * Cg;
  __shared__ float red[256];
  float sum = 0.f, sq = 0.f;
  for (int i = threadIdx.x; i < n; i += 256) {
    int hw = i / Cg, c = i - hw * Cg;
    float v = base[(size_t)hw * C + c];
    sum += v; sq += v * v;
  }
  red[threadIdx.x] = sum; __syncthreads();
  for (int s = 128; s > 0; s >>= 1) { if (threadIdx.x < s) red[threadIdx.x] += red[threadIdx.x + s]; __syncthreads(); }
  float mean = red[0] / (float)n;
  __syncthreads();
  red[threadIdx.x] = sq; __syncthreads();
  for (int s = 128; s > 0; s >>= 1) { if (threadIdx.x < s) red[threadIdx.x] += red[threadIdx.x + s]; __syncthreads(); }
  float var = red[0] / (float)n - mean * mean;
  float rstd = rsqrtf(var + 1e-6f);
  int Wp = W + 2 * pad, Hp = H + 2 * pad;
  for (int i = threadIdx.x; i < n; i += 256) {
    int hw = i / Cg, c = i - hw * Cg;
    int y = hw / W, x = hw - y * W;
    int ch = g * Cg + c;
    float v = (base[(size_t)hw * C + c] - mean) * rstd * sc[ch] + bi[ch];
    if (dosilu) v = silu_f(v);
    out[(((size_t)b * Hp + y + pad) * Wp + x + pad) * C + ch] = (f16)v;
  }
}

// ---------------- f32 NHWC -> f16 padded NHWC (no norm) ----------------
__global__ void k_cast_pad(const float* __restrict__ in, f16* __restrict__ out,
                           int B, int H, int W, int C, int pad) {
  size_t i = (size_t)blockIdx.x * blockDim.x + threadIdx.x;
  size_t total = (size_t)B * H * W * C;
  if (i >= total) return;
  int c = (int)(i % C); size_t t = i / C;
  int x = (int)(t % W); t /= W;
  int y = (int)(t % H); int b = (int)(t / H);
  out[((((size_t)b * (H + 2 * pad) + y + pad) * (W + 2 * pad)) + x + pad) * C + c] = (f16)in[i];
}

// ---------------- implicit-GEMM conv / GEMM via WMMA (templated, double-buffered) ----------------
// in: f16 [B, HP, WP, CIN] (pre-padded). wf: fragment-packed weights. Wave tile: 16(M) x 64(N).
// omode: 0 = f32 NHWC, 1 = f32 NCHW, 2 = f16 NHWC, 3 = f16 channel-major [B, N, OUTH*OUTW]
template<int KH, int KW, int CIN, int STRIDE, int OUTH, int OUTW>
__global__ void __launch_bounds__(128)
k_conv(const f16* __restrict__ in, const f16* __restrict__ wf,
       const float* __restrict__ bias, const float* __restrict__ temb,
       const float* __restrict__ addend,
       float* __restrict__ outf, f16* __restrict__ outh, int omode) {
  constexpr int N = 512;
  constexpr int NTILES = N / 16;
  constexpr int HP = OUTH * STRIDE + (KH - 1);
  constexpr int WP = OUTW * STRIDE + (KW - 1);
  constexpr int CB = CIN / 32;
  constexpr int KBLKS = KH * KW * CB;
  constexpr int PLANE = OUTH * OUTW;
  static_assert((KBLKS & 1) == 0, "KBLKS must be even for 2x unroll");

  int wave = threadIdx.x >> 5;
  int lane = threadIdx.x & 31;
  int Mtile = blockIdx.x * 64 + wave * 16;
  int ncol0 = blockIdx.y * 64;
  int nlane = lane & 15;
  int kbA = (lane >> 4) * 8;
  int nt0 = ncol0 >> 4;

  // A row for this lane
  int rowA = Mtile + nlane;
  int bA = rowA / PLANE; int rem = rowA - bA * PLANE;
  int oyA = rem / OUTW; int oxA = rem - oyA * OUTW;
  const f16* abase = in + (((size_t)bA * HP + oyA * STRIDE) * WP + oxA * STRIDE) * CIN;
  const f16* wfl = wf + ((size_t)nt0 * 32 + lane) * 16;

  CFrag acc[4];
#pragma unroll
  for (int ns = 0; ns < 4; ++ns) acc[ns].v = v8f{};

  AFrag a0, a1, b0[4], b1[4];
  auto load = [&](AFrag& a, AFrag (&bb)[4], int kblk) {
    int tap = kblk / CB;                 // CB is power of two
    int ci0 = (kblk & (CB - 1)) * 32;
    int ky = tap / KW, kx = tap - ky * KW;
    const f16* arow = abase + ((size_t)ky * WP + kx) * CIN;
    a.q[0] = *(const uint4*)(arow + ci0 + kbA);
    a.q[1] = *(const uint4*)(arow + ci0 + 16 + kbA);
    const f16* wbase = wfl + (size_t)kblk * NTILES * 512;
    __builtin_prefetch((const void*)(wbase + (size_t)2 * NTILES * 512), 0, 1);
#pragma unroll
    for (int ns = 0; ns < 4; ++ns) {
      bb[ns].q[0] = *(const uint4*)(wbase + ns * 512);
      bb[ns].q[1] = *(const uint4*)(wbase + ns * 512 + 8);
    }
  };

  load(a0, b0, 0);
#pragma unroll 1
  for (int kblk = 0; kblk < KBLKS; kblk += 2) {
    load(a1, b1, kblk + 1);
#pragma unroll
    for (int ns = 0; ns < 4; ++ns) acc[ns].v = wmma_f16(a0.v, b0[ns].v, acc[ns].v);
    if (kblk + 2 < KBLKS) load(a0, b0, kblk + 2);
#pragma unroll
    for (int ns = 0; ns < 4; ++ns) acc[ns].v = wmma_f16(a1.v, b1[ns].v, acc[ns].v);
  }

  int mh = (lane >> 4) * 8;
#pragma unroll
  for (int ns = 0; ns < 4; ++ns) {
    int col = ncol0 + ns * 16 + nlane;
    float bs = bias[col];
#pragma unroll
    for (int r = 0; r < 8; ++r) {
      int row = Mtile + mh + r;
      int b = row / PLANE; int rm = row - b * PLANE;
      int oy = rm / OUTW; int ox = rm - oy * OUTW;
      float v = acc[ns].e[r] + bs;
      if (temb)   v += temb[(size_t)b * N + col];
      if (addend) v += addend[(size_t)row * N + col];
      if (omode == 0)      outf[(size_t)row * N + col] = v;
      else if (omode == 1) outf[(((size_t)b * N + col) * OUTH + oy) * OUTW + ox] = v;
      else if (omode == 2) outh[(size_t)row * N + col] = (f16)v;
      else                 outh[((size_t)b * N + col) * PLANE + rm] = (f16)v;
    }
  }
}

// ---------------- attention: one wave per (b, head, 16-row q tile) ----------------
// q,k: f16 [B,S,C]; vT: f16 [B,C,S] (channel-major); o: f16 [B,S,C]. Dyn LDS: 16*S floats.
__global__ void __launch_bounds__(32)
k_attn(const f16* __restrict__ q, const f16* __restrict__ k, const f16* __restrict__ vT,
       f16* __restrict__ o, float scale) {
  constexpr int S = 1024, C = 512, HD = 64;
  extern __shared__ float sc[];
  int lane = threadIdx.x;
  int b = blockIdx.z, h = blockIdx.y, q0 = blockIdx.x * 16;
  int nlane = lane & 15;
  int kbA = (lane >> 4) * 8;
  int kbB = (lane >> 4) * 16;
  int mh = (lane >> 4) * 8;

  // Preload q tile A-fragments (HD=64 -> 2 K-steps of 32)
  AFrag aq[2];
  const f16* qrow = q + ((size_t)b * S + (q0 + nlane)) * C + h * HD;
#pragma unroll
  for (int ks = 0; ks < 2; ++ks) {
    aq[ks].q[0] = *(const uint4*)(qrow + ks * 32 + kbA);
    aq[ks].q[1] = *(const uint4*)(qrow + ks * 32 + 16 + kbA);
  }

  // Phase 1: scores = Q Kt
  const f16* kbase = k + ((size_t)b * S + nlane) * C + h * HD;
#pragma unroll 1
  for (int kt = 0; kt < S / 16; ++kt) {
    CFrag accs; accs.v = v8f{};
    const f16* krow = kbase + (size_t)kt * 16 * C;
#pragma unroll
    for (int ks = 0; ks < 2; ++ks) {
      AFrag bb;
      bb.q[0] = *(const uint4*)(krow + ks * 32 + kbA);
      bb.q[1] = *(const uint4*)(krow + ks * 32 + 16 + kbA);
      accs.v = wmma_f16(aq[ks].v, bb.v, accs.v);
    }
#pragma unroll
    for (int r = 0; r < 8; ++r)
      sc[(size_t)(mh + r) * S + kt * 16 + nlane] = accs.e[r] * scale;
  }
  __syncthreads();

  // Phase 2: softmax rows; f16 probs stored aliased at start of each row's f32 strip
  if (lane < 16) {
    float* row = sc + (size_t)lane * S;
    float mx = -1e30f;
    for (int i = 0; i < S; ++i) mx = fmaxf(mx, row[i]);
    float sm = 0.f;
    for (int i = 0; i < S; ++i) { float e = __expf(row[i] - mx); sm += e; row[i] = e; }
    float inv = 1.f / sm;
    f16* prow = (f16*)row;
    for (int i = 0; i < S; ++i) { float e = row[i]; prow[i] = (f16)(e * inv); }
  }
  __syncthreads();

  // Phase 3: O = P V  (HD=64 -> 4 N-subtiles; V channel-major so B-frags contiguous)
  CFrag oacc[4];
#pragma unroll
  for (int ns = 0; ns < 4; ++ns) oacc[ns].v = v8f{};
  const f16* prow = (const f16*)(sc + (size_t)nlane * S);
  const f16* vbase = vT + ((size_t)b * C + h * HD + nlane) * S;
#pragma unroll 1
  for (int kt = 0; kt < S / 32; ++kt) {
    AFrag a;
    a.q[0] = *(const uint4*)(prow + kt * 32 + kbA);
    a.q[1] = *(const uint4*)(prow + kt * 32 + 16 + kbA);
#pragma unroll
    for (int ns = 0; ns < 4; ++ns) {
      const f16* vrow = vbase + (size_t)(ns * 16) * S + kt * 32;
      AFrag bb;
      bb.q[0] = *(const uint4*)(vrow + kbA);
      bb.q[1] = *(const uint4*)(vrow + 16 + kbA);
      oacc[ns].v = wmma_f16(a.v, bb.v, oacc[ns].v);
    }
  }
#pragma unroll
  for (int ns = 0; ns < 4; ++ns)
#pragma unroll
    for (int r = 0; r < 8; ++r)
      o[((size_t)b * S + q0 + mh + r) * C + h * HD + ns * 16 + nlane] = (f16)oacc[ns].e[r];
}

// ==================== host side ====================
extern "C" void kernel_launch(void* const* d_in, const int* in_sizes, int n_in,
                              void* d_out, int out_size, void* d_ws, size_t ws_size,
                              hipStream_t stream) {
  (void)n_in; (void)out_size; (void)ws_size;
  const int B = 8, H = 32, W = 32, C1 = 256, C2 = 512, T = 512, S = 1024;

  auto F = [&](int i) { return (const float*)d_in[i]; };
  const float *X, *TEMB;
  const float *r0n1s,*r0n1b,*r0c1w,*r0c1b,*r0tw,*r0tb,*r0n2s,*r0n2b,*r0c2w,*r0c2b,*r0skw,*r0skb;
  const float *r1n1s,*r1n1b,*r1c1w,*r1c1b,*r1tw,*r1tb,*r1n2s,*r1n2b,*r1c2w,*r1c2b;
  const float *a0ns,*a0nb,*a0qw,*a0qb,*a0kw,*a0kb,*a0vw,*a0vb,*a0pw,*a0pb;
  const float *a1ns,*a1nb,*a1qw,*a1qb,*a1kw,*a1kb,*a1vw,*a1vb,*a1pw,*a1pb;
  const float *dsw,*dsb;

  bool insertion = (in_sizes[0] == B * C1 * H * W);
  if (insertion) {
    X=F(0); TEMB=F(1);
    r0n1s=F(2); r0n1b=F(3); r0c1w=F(4); r0c1b=F(5); r0tw=F(6); r0tb=F(7);
    r0n2s=F(8); r0n2b=F(9); r0c2w=F(10); r0c2b=F(11); r0skw=F(12); r0skb=F(13);
    r1n1s=F(14); r1n1b=F(15); r1c1w=F(16); r1c1b=F(17); r1tw=F(18); r1tb=F(19);
    r1n2s=F(20); r1n2b=F(21); r1c2w=F(22); r1c2b=F(23);
    a0ns=F(24); a0nb=F(25); a0qw=F(26); a0qb=F(27); a0kw=F(28); a0kb=F(29);
    a0vw=F(30); a0vb=F(31); a0pw=F(32); a0pb=F(33);
    a1ns=F(34); a1nb=F(35); a1qw=F(36); a1qb=F(37); a1kw=F(38); a1kb=F(39);
    a1vw=F(40); a1vb=F(41); a1pw=F(42); a1pb=F(43);
    dsw=F(44); dsb=F(45);
  } else { // JAX pytree (sorted-key) order
    a0kb=F(0); a0kw=F(1); a0nb=F(2); a0ns=F(3); a0pb=F(4); a0pw=F(5);
    a0qb=F(6); a0qw=F(7); a0vb=F(8); a0vw=F(9);
    a1kb=F(10); a1kw=F(11); a1nb=F(12); a1ns=F(13); a1pb=F(14); a1pw=F(15);
    a1qb=F(16); a1qw=F(17); a1vb=F(18); a1vw=F(19);
    dsb=F(20); dsw=F(21);
    r0c1b=F(22); r0c1w=F(23); r0c2b=F(24); r0c2w=F(25); r0n1b=F(26); r0n1s=F(27);
    r0n2b=F(28); r0n2s=F(29); r0skb=F(30); r0skw=F(31); r0tb=F(32); r0tw=F(33);
    r1c1b=F(34); r1c1w=F(35); r1c2b=F(36); r1c2w=F(37); r1n1b=F(38); r1n1s=F(39);
    r1n2b=F(40); r1n2s=F(41); r1tb=F(42); r1tw=F(43);
    TEMB=F(44); X=F(45);
  }

  // ---- workspace carve-out ----
  size_t off = 0;
  auto alloc = [&](size_t bytes) -> void* {
    void* p = (char*)d_ws + off;
    off += (bytes + 255) & ~(size_t)255;
    return p;
  };
  f16* wp_c1a = (f16*)alloc((size_t)9*256*512*2);
  f16* wp_c2a = (f16*)alloc((size_t)9*512*512*2);
  f16* wp_sk  = (f16*)alloc((size_t)256*512*2);
  f16* wp_c1B = (f16*)alloc((size_t)9*512*512*2);
  f16* wp_c2B = (f16*)alloc((size_t)9*512*512*2);
  f16* wp_q1 = (f16*)alloc((size_t)512*512*2);
  f16* wp_k1 = (f16*)alloc((size_t)512*512*2);
  f16* wp_v1 = (f16*)alloc((size_t)512*512*2);
  f16* wp_p1 = (f16*)alloc((size_t)512*512*2);
  f16* wp_q2 = (f16*)alloc((size_t)512*512*2);
  f16* wp_k2 = (f16*)alloc((size_t)512*512*2);
  f16* wp_v2 = (f16*)alloc((size_t)512*512*2);
  f16* wp_p2 = (f16*)alloc((size_t)512*512*2);
  f16* wp_ds = (f16*)alloc((size_t)9*512*512*2);
  float* tembA = (float*)alloc((size_t)B*512*4);
  float* tembB = (float*)alloc((size_t)B*512*4);
  float* xn  = (float*)alloc((size_t)B*S*C1*4);
  f16* x16   = (f16*)alloc((size_t)B*S*C1*2);
  size_t padBytes = (size_t)B*34*34*512*2;
  f16* padA  = (f16*)alloc(padBytes);
  f16* g16   = (f16*)alloc((size_t)B*S*C2*2);
  f16* qb16  = (f16*)alloc((size_t)B*S*C2*2);
  f16* kb16  = (f16*)alloc((size_t)B*S*C2*2);
  f16* vb16  = (f16*)alloc((size_t)B*S*C2*2);   // V transposed [B,C,S]
  float* bufA = (float*)alloc((size_t)B*S*C2*4);
  float* bufB = (float*)alloc((size_t)B*S*C2*4);

  auto packw = [&](const float* w, f16* dst, int Cout, int Cin, int taps) {
    size_t total = (size_t)Cout * Cin * taps;
    k_pack_wf<<<(unsigned)((total + 255) / 256), 256, 0, stream>>>(w, dst, Cout, Cin, taps);
  };
  packw(r0c1w, wp_c1a, 512, 256, 9);  packw(r0c2w, wp_c2a, 512, 512, 9);
  packw(r0skw, wp_sk, 512, 256, 1);
  packw(r1c1w, wp_c1B, 512, 512, 9);  packw(r1c2w, wp_c2B, 512, 512, 9);
  packw(a0qw, wp_q1, 512, 512, 1);    packw(a0kw, wp_k1, 512, 512, 1);
  packw(a0vw, wp_v1, 512, 512, 1);    packw(a0pw, wp_p1, 512, 512, 1);
  packw(a1qw, wp_q2, 512, 512, 1);    packw(a1kw, wp_k2, 512, 512, 1);
  packw(a1vw, wp_v2, 512, 512, 1);    packw(a1pw, wp_p2, 512, 512, 1);
  packw(dsw, wp_ds, 512, 512, 9);

  k_temb<<<dim3(2, B), 256, 0, stream>>>(TEMB, r0tw, r0tb, tembA, T, 512);
  k_temb<<<dim3(2, B), 256, 0, stream>>>(TEMB, r1tw, r1tb, tembB, T, 512);

  {
    size_t tot = (size_t)B * C1 * S;
    k_nchw_to_nhwc<<<(unsigned)((tot + 255) / 256), 256, 0, stream>>>(X, xn, x16, B, C1, S);
  }

  dim3 convGrid(128, 8), convBlk(128);

  // ---- ResNet block 1 ----
  hipMemsetAsync(padA, 0, padBytes, stream);
  k_gn<<<dim3(B, 32), 256, 0, stream>>>(xn, r0n1s, r0n1b, padA, H, W, 256, 8, 1, 1);
  k_conv<3,3,256,1,32,32><<<convGrid, convBlk, 0, stream>>>(padA, wp_c1a,
      r0c1b, tembA, nullptr, bufA, nullptr, 0);
  hipMemsetAsync(padA, 0, padBytes, stream);
  k_gn<<<dim3(B, 32), 256, 0, stream>>>(bufA, r0n2s, r0n2b, padA, H, W, 512, 16, 1, 1);
  k_conv<3,3,512,1,32,32><<<convGrid, convBlk, 0, stream>>>(padA, wp_c2a,
      r0c2b, nullptr, nullptr, bufB, nullptr, 0);
  k_conv<1,1,256,1,32,32><<<convGrid, convBlk, 0, stream>>>(x16, wp_sk,
      r0skb, nullptr, bufB, bufA, nullptr, 0);                      // bufA = res1_out

  // ---- Attention block 1 ----
  k_gn<<<dim3(B, 32), 256, 0, stream>>>(bufA, a0ns, a0nb, g16, H, W, 512, 16, 0, 0);
  k_conv<1,1,512,1,32,32><<<convGrid, convBlk, 0, stream>>>(g16, wp_q1,
      a0qb, nullptr, nullptr, nullptr, qb16, 2);
  k_conv<1,1,512,1,32,32><<<convGrid, convBlk, 0, stream>>>(g16, wp_k1,
      a0kb, nullptr, nullptr, nullptr, kb16, 2);
  k_conv<1,1,512,1,32,32><<<convGrid, convBlk, 0, stream>>>(g16, wp_v1,
      a0vb, nullptr, nullptr, nullptr, vb16, 3);                    // V transposed
  k_attn<<<dim3(S / 16, 8, B), 32, 16 * S * sizeof(float), stream>>>(
      qb16, kb16, vb16, g16, 0.125f);
  k_conv<1,1,512,1,32,32><<<convGrid, convBlk, 0, stream>>>(g16, wp_p1,
      a0pb, nullptr, bufA, bufB, nullptr, 0);                       // bufB = attn1_out

  // ---- ResNet block 2 ----
  hipMemsetAsync(padA, 0, padBytes, stream);
  k_gn<<<dim3(B, 32), 256, 0, stream>>>(bufB, r1n1s, r1n1b, padA, H, W, 512, 16, 1, 1);
  k_conv<3,3,512,1,32,32><<<convGrid, convBlk, 0, stream>>>(padA, wp_c1B,
      r1c1b, tembB, nullptr, bufA, nullptr, 0);
  hipMemsetAsync(padA, 0, padBytes, stream);
  k_gn<<<dim3(B, 32), 256, 0, stream>>>(bufA, r1n2s, r1n2b, padA, H, W, 512, 16, 1, 1);
  k_conv<3,3,512,1,32,32><<<convGrid, convBlk, 0, stream>>>(padA, wp_c2B,
      r1c2b, nullptr, bufB, bufA, nullptr, 0);                      // bufA = res2_out

  // ---- Attention block 2 ----
  k_gn<<<dim3(B, 32), 256, 0, stream>>>(bufA, a1ns, a1nb, g16, H, W, 512, 16, 0, 0);
  k_conv<1,1,512,1,32,32><<<convGrid, convBlk, 0, stream>>>(g16, wp_q2,
      a1qb, nullptr, nullptr, nullptr, qb16, 2);
  k_conv<1,1,512,1,32,32><<<convGrid, convBlk, 0, stream>>>(g16, wp_k2,
      a1kb, nullptr, nullptr, nullptr, kb16, 2);
  k_conv<1,1,512,1,32,32><<<convGrid, convBlk, 0, stream>>>(g16, wp_v2,
      a1vb, nullptr, nullptr, nullptr, vb16, 3);                    // V transposed
  k_attn<<<dim3(S / 16, 8, B), 32, 16 * S * sizeof(float), stream>>>(
      qb16, kb16, vb16, g16, 0.125f);
  k_conv<1,1,512,1,32,32><<<convGrid, convBlk, 0, stream>>>(g16, wp_p2,
      a1pb, nullptr, bufA, bufB, nullptr, 0);                       // bufB = attn2_out

  // ---- Downsample: stride-2 3x3 conv, output f32 NCHW -> d_out ----
  hipMemsetAsync(padA, 0, padBytes, stream);
  {
    size_t tot = (size_t)B * S * C2;
    k_cast_pad<<<(unsigned)((tot + 255) / 256), 256, 0, stream>>>(bufB, padA, B, H, W, 512, 1);
  }
  k_conv<3,3,512,2,16,16><<<dim3(32, 8), convBlk, 0, stream>>>(padA, wp_ds,
      dsb, nullptr, nullptr, (float*)d_out, nullptr, 1);
}
